// FloorPlanGAT_82325933130370
// MI455X (gfx1250) — matrix operand order repaired and checked by reading, not compiled, not run on hardware
//
#include <hip/hip_runtime.h>
#include <cstring>

#define NN   50000
#define EE   300000
#define FIN  32
#define HHN  4
#define CCN  64
#define HIDN 256
#define EDIM 7
#define GGN  64
#define ETOT (EE + NN)

typedef __attribute__((ext_vector_type(16))) __bf16 v16bf;
typedef __attribute__((ext_vector_type(8)))  __bf16 v8bf;
typedef __attribute__((ext_vector_type(8)))  float  v8f;

// ---- order-preserving float <-> uint encoding for atomic max ----
__device__ __forceinline__ unsigned fenc(float f) {
  unsigned u = __float_as_uint(f);
  return (u & 0x80000000u) ? ~u : (u | 0x80000000u);
}
__device__ __forceinline__ float fdec(unsigned u) {
  return (u & 0x80000000u) ? __uint_as_float(u & 0x7FFFFFFFu) : __uint_as_float(~u);
}

__global__ void fill_u32(unsigned* p, unsigned v, int n) {
  int i = blockIdx.x * blockDim.x + threadIdx.x;
  if (i < n) p[i] = v;
}

// degree count + per-dst edge_attr sums (for self-loop fill_value='mean')
__global__ void deg_kernel(const int* __restrict__ ei, const float* __restrict__ ea,
                           float* __restrict__ loop_sum, float* __restrict__ cnt) {
  int e = blockIdx.x * blockDim.x + threadIdx.x;
  if (e >= EE) return;
  int d = ei[EE + e];
  atomicAdd(&cnt[d], 1.0f);
#pragma unroll
  for (int k = 0; k < EDIM; k++) atomicAdd(&loop_sum[d * EDIM + k], ea[e * EDIM + k]);
}

__global__ void loop_fin(float* __restrict__ loop_attr, const float* __restrict__ cnt) {
  int i = blockIdx.x * blockDim.x + threadIdx.x;
  if (i >= NN * EDIM) return;
  loop_attr[i] = loop_attr[i] / fmaxf(cnt[i / EDIM], 1.0f);
}

// h = x @ in_W + in_b   [N,32]x[32,256]
__global__ void inproj(const float* __restrict__ x, const float* __restrict__ W,
                       const float* __restrict__ b, float* __restrict__ h) {
  int t = blockIdx.x * blockDim.x + threadIdx.x;
  if (t >= NN * HIDN) return;
  int n = t >> 8, j = t & 255;
  const float* xr = &x[n * FIN];
  float acc = b[j];
#pragma unroll
  for (int k = 0; k < FIN; k++) acc += xr[k] * W[k * HIDN + j];
  h[t] = acc;
}

__global__ void cvt_bf16(const float* __restrict__ s, __bf16* __restrict__ d, int n) {
  int i = blockIdx.x * blockDim.x + threadIdx.x;
  if (i < n) d[i] = (__bf16)s[i];
}

// Wt[n][k] = bf16(W[k][n])  (transpose so B loads are contiguous per column)
__global__ void prep_wt(const float* __restrict__ W, __bf16* __restrict__ Wt) {
  int t = blockIdx.x * blockDim.x + threadIdx.x;
  if (t >= HIDN * HIDN) return;
  int k = t >> 8, n = t & 255;
  Wt[n * HIDN + k] = (__bf16)W[k * HIDN + n];
}

// we_att[k][hh] = sum_c W_e[k, hh*64+c] * att_edge[hh*64+c]
__global__ void we_att_k(const float* __restrict__ W_e, const float* __restrict__ att_e,
                         float* __restrict__ we_att) {
  int t = threadIdx.x;
  if (t >= EDIM * HHN) return;
  int k = t / HHN, hh = t % HHN;
  float s = 0.f;
  for (int c = 0; c < CCN; c++) s += W_e[k * HIDN + hh * CCN + c] * att_e[hh * CCN + c];
  we_att[t] = s;
}

// xh = A(bf16)[M,256] @ W via Wt (B col-major copy), f32 accumulate. One wave per 16x16 tile.
__global__ void __launch_bounds__(256) gemm_wmma(const __bf16* __restrict__ A,
                                                 const __bf16* __restrict__ B,
                                                 float* __restrict__ C, int M) {
  int w = threadIdx.x >> 5, lane = threadIdx.x & 31;
  int tile = blockIdx.x * 8 + w;
  int tn = tile & 15, tm = tile >> 4;
  int m0 = tm * 16, n0 = tn * 16;
  if (m0 >= M) return;                         // wave-uniform
  int half = lane >> 4, l15 = lane & 15;
  const __bf16* arow = A + (size_t)(m0 + l15) * HIDN;
  const __bf16* brow = B + (size_t)(n0 + l15) * HIDN;
  v8f c = {};
#pragma unroll
  for (int kb = 0; kb < 8; kb++) {
    int k0 = kb * 32;
    // A 16x32 bf16 layout: lane-half holds K = k0 + half*8 + {0..7}, then +16
    v8bf alo = *(const v8bf*)(arow + k0 + half * 8);
    v8bf ahi = *(const v8bf*)(arow + k0 + 16 + half * 8);
    // B 32x16 layout: lane-half holds K = k0 + half*8 + {0..7}, then +16
    v8bf blo = *(const v8bf*)(brow + k0 + half * 8);
    v8bf bhi = *(const v8bf*)(brow + k0 + 16 + half * 8);
    if (kb < 7) __builtin_prefetch(arow + k0 + 32, 0, 3);
    v16bf a, b;
#pragma unroll
    for (int i = 0; i < 8; i++) { a[i] = alo[i]; a[i + 8] = ahi[i]; b[i] = blo[i]; b[i + 8] = bhi[i]; }
    c = __builtin_amdgcn_wmma_f32_16x16x32_bf16(false, a, false, b, (short)0, c, false, false);
  }
  float* crow = C + (size_t)(m0 + 8 * half) * HIDN + n0 + l15;
#pragma unroll
  for (int v = 0; v < 8; v++) crow[(size_t)v * HIDN] = c[v];
}

// per-node attention scalars
__global__ void attn_node(const float* __restrict__ xh, const float* __restrict__ as,
                          const float* __restrict__ ad, float* __restrict__ a_src,
                          float* __restrict__ a_dst) {
  int t = blockIdx.x * blockDim.x + threadIdx.x;
  if (t >= NN * HHN) return;
  int node = t >> 2, hh = t & 3;
  const float* xr = &xh[(size_t)node * HIDN + hh * CCN];
  float s = 0.f, d = 0.f;
  for (int c = 0; c < CCN; c++) { float v = xr[c]; s += v * as[hh * CCN + c]; d += v * ad[hh * CCN + c]; }
  a_src[t] = s; a_dst[t] = d;
}

__global__ void alpha_k(const int* __restrict__ ei, const float* __restrict__ ea,
                        const float* __restrict__ loop_attr, const float* __restrict__ we_att,
                        const float* __restrict__ a_src, const float* __restrict__ a_dst,
                        float* __restrict__ ebuf, unsigned* __restrict__ amax) {
  int e = blockIdx.x * blockDim.x + threadIdx.x;
  if (e >= ETOT) return;
  int s, d; const float* eap;
  if (e < EE) { s = ei[e]; d = ei[EE + e]; eap = &ea[(size_t)e * EDIM]; }
  else        { s = e - EE; d = s;         eap = &loop_attr[(size_t)(e - EE) * EDIM]; }
  float eav[EDIM];
#pragma unroll
  for (int k = 0; k < EDIM; k++) eav[k] = eap[k];
#pragma unroll
  for (int hh = 0; hh < HHN; hh++) {
    float aev = 0.f;
#pragma unroll
    for (int k = 0; k < EDIM; k++) aev += eav[k] * we_att[k * HHN + hh];
    float al = a_src[s * HHN + hh] + a_dst[d * HHN + hh] + aev;
    al = al >= 0.f ? al : 0.2f * al;          // LeakyReLU(0.2)
    ebuf[(size_t)e * HHN + hh] = al;
    atomicMax(&amax[d * HHN + hh], fenc(al));
  }
}

__global__ void ex_k(const int* __restrict__ ei, float* __restrict__ ebuf,
                     const unsigned* __restrict__ amax, float* __restrict__ den) {
  int e = blockIdx.x * blockDim.x + threadIdx.x;
  if (e >= ETOT) return;
  int d = (e < EE) ? ei[EE + e] : e - EE;
#pragma unroll
  for (int hh = 0; hh < HHN; hh++) {
    float mx = fdec(amax[d * HHN + hh]);
    float ex = __expf(ebuf[(size_t)e * HHN + hh] - mx);
    ebuf[(size_t)e * HHN + hh] = ex;
    atomicAdd(&den[d * HHN + hh], ex);
  }
}

// message pass: one lane per (edge, 8 channels), native f32 scatter-add
__global__ void msg_k(const int* __restrict__ ei, const float* __restrict__ xh,
                      const float* __restrict__ ebuf, const float* __restrict__ den,
                      float* __restrict__ outb) {
  long long t = (long long)blockIdx.x * blockDim.x + threadIdx.x;
  if (t >= (long long)ETOT * 32) return;
  int e = (int)(t >> 5), lane = (int)(t & 31);
  int s, d;
  if (e < EE) { s = ei[e]; d = ei[EE + e]; } else { s = e - EE; d = s; }
  int cb = lane * 8, hh = cb >> 6;
  float att = ebuf[(size_t)e * HHN + hh] / den[d * HHN + hh];
  const float* xr = &xh[(size_t)s * HIDN + cb];
  float* orow = &outb[(size_t)d * HIDN + cb];
#pragma unroll
  for (int i = 0; i < 8; i++) atomicAdd(&orow[i], xr[i] * att);
}

// bias + layernorm + elu + residual (in-place on h); one 256-thread block per node
__global__ void __launch_bounds__(256) post_k(const float* __restrict__ conv,
                                              const float* __restrict__ bias,
                                              const float* __restrict__ lg,
                                              const float* __restrict__ lb,
                                              float* __restrict__ h) {
  __shared__ float red[256];
  int node = blockIdx.x, c = threadIdx.x;
  float v = conv[(size_t)node * HIDN + c] + bias[c];
  red[c] = v; __syncthreads();
  for (int s = 128; s > 0; s >>= 1) { if (c < s) red[c] += red[c + s]; __syncthreads(); }
  float mu = red[0] * (1.0f / HIDN); __syncthreads();
  float dv = v - mu;
  red[c] = dv * dv; __syncthreads();
  for (int s = 128; s > 0; s >>= 1) { if (c < s) red[c] += red[c + s]; __syncthreads(); }
  float var = red[0] * (1.0f / HIDN);
  float y = dv * rsqrtf(var + 1e-5f) * lg[c] + lb[c];
  y = y > 0.f ? y : (__expf(y) - 1.f);        // ELU
  h[(size_t)node * HIDN + c] = y + h[(size_t)node * HIDN + c];
}

__global__ void pool_k(const float* __restrict__ h, const int* __restrict__ fm,
                       const int* __restrict__ batch, float* __restrict__ sums,
                       float* __restrict__ cnts, unsigned* __restrict__ maxenc) {
  int t = blockIdx.x * blockDim.x + threadIdx.x;
  if (t >= NN * HIDN) return;
  int node = t >> 8, c = t & 255;
  int m = fm[node] & 1, g = batch[node];
  float v = h[t];
  int base = (m * GGN + g) * HIDN + c;
  atomicAdd(&sums[base], v);
  atomicMax(&maxenc[base], fenc(v));
  if (c == 0) atomicAdd(&cnts[m * GGN + g], 1.0f);
}

__global__ void repr_k(const float* __restrict__ sums, const float* __restrict__ cnts,
                       const unsigned* __restrict__ maxenc, float* __restrict__ gr) {
  int t = blockIdx.x * blockDim.x + threadIdx.x;
  if (t >= GGN * 1024) return;
  int g = t >> 10, j = t & 1023;
  int m = j >> 9, jj = j & 511;
  float out;
  if (jj < HIDN) out = sums[(m * GGN + g) * HIDN + jj] / fmaxf(cnts[m * GGN + g], 1.0f);
  else {
    float mx = fdec(maxenc[(m * GGN + g) * HIDN + (jj - HIDN)]);
    out = (mx <= -5e29f) ? 0.f : mx;
  }
  gr[t] = out;
}

__global__ void __launch_bounds__(128) head_k(const float* __restrict__ gr,
                                              const float* __restrict__ W1,
                                              const float* __restrict__ b1,
                                              const float* __restrict__ W2,
                                              const float* __restrict__ b2,
                                              float* __restrict__ out) {
  __shared__ float red[128];
  int g = blockIdx.x, j = threadIdx.x;
  const float* gp = &gr[g * 1024];
  float acc = b1[j];
  for (int i = 0; i < 1024; i++) acc += gp[i] * W1[i * 128 + j];
  float z = fmaxf(acc, 0.f);
  red[j] = z * W2[j]; __syncthreads();
  for (int s = 64; s > 0; s >>= 1) { if (j < s) red[j] += red[j + s]; __syncthreads(); }
  if (j == 0) { float zz = red[0] + b2[0]; out[g] = 1.f / (1.f + __expf(-zz)); }
}

static unsigned host_fenc(float f) {
  unsigned u; memcpy(&u, &f, 4);
  return (u & 0x80000000u) ? ~u : (u | 0x80000000u);
}

struct LayerP { const float *W, *att_src, *att_dst, *W_e, *att_edge, *bias, *ln_g, *ln_b; };
struct HeadP  { const float *W1, *b1, *W2, *b2; };

extern "C" void kernel_launch(void* const* d_in, const int* in_sizes, int n_in,
                              void* d_out, int out_size, void* d_ws, size_t ws_size,
                              hipStream_t stream) {
  const float* x     = (const float*)d_in[0];
  const int*   ei    = (const int*)d_in[1];
  const float* ea    = (const float*)d_in[2];
  const int*   fm    = (const int*)d_in[3];
  const int*   batch = (const int*)d_in[4];

  const float *in_W, *in_b; LayerP L[3]; HeadP Hd[5];
  auto F = [&](int i) { return (const float*)d_in[i]; };
  if (in_sizes[5] == FIN * HIDN) {
    // insertion-order flattening: in_W, in_b, layers(W,att_src,att_dst,W_e,att_edge,bias,ln_g,ln_b), heads(W1,b1,W2,b2)
    in_W = F(5); in_b = F(6);
    int idx = 7;
    for (int i = 0; i < 3; i++) { L[i].W=F(idx++); L[i].att_src=F(idx++); L[i].att_dst=F(idx++);
      L[i].W_e=F(idx++); L[i].att_edge=F(idx++); L[i].bias=F(idx++); L[i].ln_g=F(idx++); L[i].ln_b=F(idx++); }
    for (int j = 0; j < 5; j++) { Hd[j].W1=F(idx++); Hd[j].b1=F(idx++); Hd[j].W2=F(idx++); Hd[j].b2=F(idx++); }
  } else {
    // jax sorted-pytree flattening: heads(W1,W2,b1,b2)x5, in_W, in_b, layers(W,W_e,att_dst,att_edge,att_src,bias,ln_b,ln_g)x3
    int idx = 5;
    for (int j = 0; j < 5; j++) { Hd[j].W1=F(idx++); Hd[j].W2=F(idx++); Hd[j].b1=F(idx++); Hd[j].b2=F(idx++); }
    in_W = F(idx++); in_b = F(idx++);
    for (int i = 0; i < 3; i++) { L[i].W=F(idx++); L[i].W_e=F(idx++); L[i].att_dst=F(idx++);
      L[i].att_edge=F(idx++); L[i].att_src=F(idx++); L[i].bias=F(idx++); L[i].ln_b=F(idx++); L[i].ln_g=F(idx++); }
  }

  // ---- workspace carve (256B aligned) ----
  char* base = (char*)d_ws; size_t off = 0;
  auto carve = [&](size_t bytes) -> void* { void* r = base + off; off = (off + bytes + 255) & ~(size_t)255; return r; };
  float*    h      = (float*)   carve((size_t)NN * HIDN * 4);
  float*    conv   = (float*)   carve((size_t)NN * HIDN * 4);
  float*    xh     = (float*)   carve((size_t)NN * HIDN * 4);
  __bf16*   hbf    = (__bf16*)  carve((size_t)NN * HIDN * 2);
  __bf16*   Wt     = (__bf16*)  carve((size_t)HIDN * HIDN * 2);
  float*    a_src  = (float*)   carve((size_t)NN * HHN * 4);
  float*    a_dst  = (float*)   carve((size_t)NN * HHN * 4);
  float*    loopat = (float*)   carve((size_t)NN * EDIM * 4);
  float*    cnt    = (float*)   carve((size_t)NN * 4);
  unsigned* amax   = (unsigned*)carve((size_t)NN * HHN * 4);
  float*    den    = (float*)   carve((size_t)NN * HHN * 4);
  float*    ebuf   = (float*)   carve((size_t)ETOT * HHN * 4);
  float*    weatt  = (float*)   carve(EDIM * HHN * 4);
  float*    sums   = (float*)   carve(2 * GGN * HIDN * 4);
  float*    cnts   = (float*)   carve(2 * GGN * 4);
  unsigned* maxenc = (unsigned*)carve(2 * GGN * HIDN * 4);
  float*    gr     = (float*)   carve(GGN * 1024 * 4);

  const unsigned NEG = host_fenc(-1e30f);
  auto blks = [](long long n, int b) { return (unsigned)((n + b - 1) / b); };

  // ---- self-loop mean edge_attr ----
  fill_u32<<<blks(NN, 256), 256, 0, stream>>>((unsigned*)cnt, 0u, NN);
  fill_u32<<<blks((long long)NN * EDIM, 256), 256, 0, stream>>>((unsigned*)loopat, 0u, NN * EDIM);
  deg_kernel<<<blks(EE, 256), 256, 0, stream>>>(ei, ea, loopat, cnt);
  loop_fin<<<blks((long long)NN * EDIM, 256), 256, 0, stream>>>(loopat, cnt);

  // ---- input projection ----
  inproj<<<blks((long long)NN * HIDN, 256), 256, 0, stream>>>(x, in_W, in_b, h);

  // ---- 3 GATConv layers ----
  for (int l = 0; l < 3; l++) {
    prep_wt<<<blks(HIDN * HIDN, 256), 256, 0, stream>>>(L[l].W, Wt);
    we_att_k<<<1, 32, 0, stream>>>(L[l].W_e, L[l].att_edge, weatt);
    cvt_bf16<<<blks((long long)NN * HIDN, 256), 256, 0, stream>>>(h, hbf, NN * HIDN);
    gemm_wmma<<<(NN / 16) * 16 / 8, 256, 0, stream>>>(hbf, Wt, xh, NN);
    attn_node<<<blks((long long)NN * HHN, 256), 256, 0, stream>>>(xh, L[l].att_src, L[l].att_dst, a_src, a_dst);
    fill_u32<<<blks((long long)NN * HHN, 256), 256, 0, stream>>>(amax, NEG, NN * HHN);
    fill_u32<<<blks((long long)NN * HHN, 256), 256, 0, stream>>>((unsigned*)den, 0u, NN * HHN);
    alpha_k<<<blks(ETOT, 256), 256, 0, stream>>>(ei, ea, loopat, weatt, a_src, a_dst, ebuf, amax);
    ex_k<<<blks(ETOT, 256), 256, 0, stream>>>(ei, ebuf, amax, den);
    fill_u32<<<blks((long long)NN * HIDN, 256), 256, 0, stream>>>((unsigned*)conv, 0u, NN * HIDN);
    msg_k<<<blks((long long)ETOT * 32, 256), 256, 0, stream>>>(ei, xh, ebuf, den, conv);
    post_k<<<NN, 256, 0, stream>>>(conv, L[l].bias, L[l].ln_g, L[l].ln_b, h);
  }

  // ---- dual masked pooling (mean || max per mask) ----
  fill_u32<<<blks(2 * GGN * HIDN, 256), 256, 0, stream>>>((unsigned*)sums, 0u, 2 * GGN * HIDN);
  fill_u32<<<1, 256, 0, stream>>>((unsigned*)cnts, 0u, 2 * GGN);
  fill_u32<<<blks(2 * GGN * HIDN, 256), 256, 0, stream>>>(maxenc, NEG, 2 * GGN * HIDN);
  pool_k<<<blks((long long)NN * HIDN, 256), 256, 0, stream>>>(h, fm, batch, sums, cnts, maxenc);
  repr_k<<<blks(GGN * 1024, 256), 256, 0, stream>>>(sums, cnts, maxenc, gr);

  // ---- 5 prediction heads ----
  float* outp = (float*)d_out;
  for (int hd = 0; hd < 5; hd++)
    head_k<<<GGN, 128, 0, stream>>>(gr, Hd[hd].W1, Hd[hd].b1, Hd[hd].W2, Hd[hd].b2, outp + hd * GGN);
}
